// SelfAttentionLayer_62886911148688
// MI455X (gfx1250) — matrix operand order repaired and checked by reading, not compiled, not run on hardware
//
#include <hip/hip_runtime.h>
#include <math.h>

// ---------------------------------------------------------------------------
// Transformer-XL relative-attention block for MI455X (gfx1250, wave32, WMMA)
// All GEMMs run through v_wmma_f32_16x16x32_bf16 (f32 accumulate).
// ---------------------------------------------------------------------------

typedef __bf16 bf16;
typedef __attribute__((ext_vector_type(16))) __bf16 v16bf;
typedef __attribute__((ext_vector_type(8)))  __bf16 v8bf;
typedef __attribute__((ext_vector_type(8)))  float  v8f;

#define HDIM 512
#define LCTX 2048
#define NB   4
#define NH   8
#define DH   64

enum { OUT_BF16 = 0, OUT_F32 = 1, OUT_GELU_BF16 = 2, OUT_BF16_VT = 3 };

__device__ __forceinline__ v8f wmma_bf16(v16bf a, v16bf b, v8f c) {
  return __builtin_amdgcn_wmma_f32_16x16x32_bf16(false, a, false, b, (short)0, c,
                                                 false, false);
}
__device__ __forceinline__ v8f zero_v8f() {
  v8f z = {0.f, 0.f, 0.f, 0.f, 0.f, 0.f, 0.f, 0.f};
  return z;
}
// A-operand (16x32 bf16): lane l<16 = row l, K0..15; lane l+16 = K16..31
__device__ __forceinline__ v16bf load_matA(const bf16* rowp, int hf) {
  union { v16bf v; v8bf h[2]; } u;
  u.h[0] = *(const v8bf*)(rowp + (hf << 4));
  u.h[1] = *(const v8bf*)(rowp + (hf << 4) + 8);
  return u.v;
}
// B-operand (32x16 bf16): lane n<16 = col n rows K0..15; lane n+16 = K16..31
__device__ __forceinline__ v16bf load_matB(const bf16* colp, int hf) {
  return *(const v16bf*)(colp + (hf << 4));
}

// ---------------------------------------------------------------------------
// Generic bf16 WMMA GEMM: C[M x 512] = A[M x 512] * Bw^T + bias, epilogue modes.
// Bw stored row-major as [out_col][K] so B-operand loads are contiguous.
// Block = 128 threads (4 waves); each wave owns a 32x64 output tile.
// ---------------------------------------------------------------------------
__global__ __launch_bounds__(128) void gemm_bf16_wmma(
    const bf16* __restrict__ A, const bf16* __restrict__ Bw,
    const float* __restrict__ bias, void* __restrict__ Cout, int M, int mode)
{
  const int lane = threadIdx.x & 31;
  const int wave = threadIdx.x >> 5;
  const int hf   = lane >> 4;
  const int ln   = lane & 15;
  const int m0   = blockIdx.x * 128 + wave * 32;
  const int n0   = blockIdx.y * 64;
  (void)M;

  v8f acc[2][4];
#pragma unroll
  for (int rt = 0; rt < 2; ++rt)
#pragma unroll
    for (int nt = 0; nt < 4; ++nt) acc[rt][nt] = zero_v8f();

  const bf16* arow0 = A + (size_t)(m0 + ln) * HDIM;
  const bf16* arow1 = A + (size_t)(m0 + 16 + ln) * HDIM;

  for (int kk = 0; kk < HDIM; kk += 32) {
    __builtin_prefetch(arow0 + kk + 64, 0, 1);   // global_prefetch_b8 next k-slab
    v16bf a0 = load_matA(arow0 + kk, hf);
    v16bf a1 = load_matA(arow1 + kk, hf);
    v16bf bt[4];
#pragma unroll
    for (int nt = 0; nt < 4; ++nt)
      bt[nt] = load_matB(Bw + (size_t)(n0 + nt * 16 + ln) * HDIM + kk, hf);
#pragma unroll
    for (int nt = 0; nt < 4; ++nt) {
      acc[0][nt] = wmma_bf16(a0, bt[nt], acc[0][nt]);
      acc[1][nt] = wmma_bf16(a1, bt[nt], acc[1][nt]);
    }
  }

#pragma unroll
  for (int rt = 0; rt < 2; ++rt) {
#pragma unroll
    for (int nt = 0; nt < 4; ++nt) {
      const int col = n0 + nt * 16 + ln;
      const float bv = bias ? bias[col] : 0.f;
#pragma unroll
      for (int v = 0; v < 8; ++v) {
        const int row = m0 + rt * 16 + v + 8 * hf;
        float val = acc[rt][nt][v] + bv;
        if (mode == OUT_GELU_BF16)
          val = 0.5f * val * (1.f + erff(val * 0.70710678118654752f));
        if (mode == OUT_F32) {
          ((float*)Cout)[(size_t)row * HDIM + col] = val;
        } else if (mode == OUT_BF16_VT) {
          // store v transposed per head: vT[(b*NH+n)*DH + d][l]
          const int bb = row >> 11, l = row & (LCTX - 1);
          const int n = col >> 6, d = col & 63;
          ((bf16*)Cout)[(size_t)((bb * NH + n) * DH + d) * LCTX + l] = (bf16)val;
        } else {
          ((bf16*)Cout)[(size_t)row * HDIM + col] = (bf16)val;
        }
      }
    }
  }
}

// ---------------------------------------------------------------------------
// Relative-position causal attention, one wave per (b, n, 32-query block).
// Online (flash) softmax; rel_shift realized as diagonal LDS gather of the
// positional WMMA slab:  shifted[i,j] = q[i] . r[L-1-i+j]  (valid for j <= i).
// ---------------------------------------------------------------------------
__global__ __launch_bounds__(32) void rel_attention_wmma(
    const bf16* __restrict__ q, const bf16* __restrict__ k,
    const bf16* __restrict__ vT, const bf16* __restrict__ r,
    const float* __restrict__ ckb, const float* __restrict__ prb,
    bf16* __restrict__ attn_out)
{
  __shared__ __align__(32) float slab[16 * 32];      // positional score slab
  __shared__ __align__(32) bf16  wP[2][16 * 32];     // probabilities (A layout)

  const int lane = threadIdx.x;
  const int hf = lane >> 4;
  const int ln = lane & 15;
  const int qb = blockIdx.x;
  const int bn = blockIdx.y;
  const int b = bn >> 3, n = bn & 7;
  const int i0 = qb * 32;

  // q rows stay resident in VGPRs for the whole kernel
  v16bf qA[2][2];
#pragma unroll
  for (int rt = 0; rt < 2; ++rt) {
    const bf16* qrow = q + (size_t)(b * LCTX + i0 + rt * 16 + ln) * HDIM + n * DH;
    qA[rt][0] = load_matA(qrow, hf);
    qA[rt][1] = load_matA(qrow + 32, hf);
  }

  float mrow[2][8], lrow[2][8];
  v8f O[2][4];
#pragma unroll
  for (int rt = 0; rt < 2; ++rt) {
#pragma unroll
    for (int v = 0; v < 8; ++v) { mrow[rt][v] = -1e30f; lrow[rt][v] = 0.f; }
#pragma unroll
    for (int nt = 0; nt < 4; ++nt) O[rt][nt] = zero_v8f();
  }

  const int nkb = qb + 1;   // causal: key blocks 0..qb
  for (int kb = 0; kb < nkb; ++kb) {
    const int j0 = kb * 32;

    v16bf kB[2][2];
#pragma unroll
    for (int js = 0; js < 2; ++js) {
      const bf16* krow = k + (size_t)(b * LCTX + j0 + js * 16 + ln) * HDIM + n * DH;
      kB[js][0] = load_matB(krow, hf);
      kB[js][1] = load_matB(krow + 32, hf);
    }

#pragma unroll
    for (int rt = 0; rt < 2; ++rt) {
      float Sv[2][8];
#pragma unroll
      for (int js = 0; js < 2; ++js) {
        v8f S = zero_v8f();
        S = wmma_bf16(qA[rt][0], kB[js][0], S);
        S = wmma_bf16(qA[rt][1], kB[js][1], S);

        // positional slab: p in [tb-16, tb+16)
        const int tb = (LCTX - 1) - (i0 + rt * 16) + (j0 + js * 16);
        __syncthreads();   // protect slab WAR vs previous extraction
#pragma unroll
        for (int half = 0; half < 2; ++half) {
          const int pb = tb - 16 + half * 16;
          int pc = pb + ln;
          pc = pc < 0 ? 0 : (pc > LCTX - 1 ? LCTX - 1 : pc);
          const bf16* rrow = r + (size_t)pc * HDIM + n * DH;
          v8f P = zero_v8f();
          P = wmma_bf16(qA[rt][0], load_matB(rrow, hf), P);
          P = wmma_bf16(qA[rt][1], load_matB(rrow + 32, hf), P);
          const int pg = pb + ln;
          const float prbv = (pg >= 0 && pg < LCTX) ? prb[n * LCTX + pg] : 0.f;
#pragma unroll
          for (int v = 0; v < 8; ++v)
            slab[(v + 8 * hf) * 32 + half * 16 + ln] = P[v] + prbv;
        }
        __syncthreads();

        const int jg = j0 + js * 16 + ln;
        const float ckv = ckb[(size_t)bn * LCTX + jg];
#pragma unroll
        for (int v = 0; v < 8; ++v) {
          const int ii = v + 8 * hf;
          float s = S[v] + slab[ii * 32 + 16 + ln - ii];   // rel_shift gather
          s = (s + ckv) * 0.125f;                          // 1/sqrt(64)
          const int ig = i0 + rt * 16 + ii;
          if (jg > ig) s = -3.0e38f;                       // causal mask
          Sv[js][v] = s;
        }
      }

      // online softmax over this 32-key block (half-wave = one row group)
#pragma unroll
      for (int v = 0; v < 8; ++v) {
        float mr = fmaxf(Sv[0][v], Sv[1][v]);
#pragma unroll
        for (int off = 8; off > 0; off >>= 1) mr = fmaxf(mr, __shfl_xor(mr, off, 32));
        const float mn = fmaxf(mrow[rt][v], mr);
        const float sc = __expf(mrow[rt][v] - mn);
        const float p0 = __expf(Sv[0][v] - mn);
        const float p1 = __expf(Sv[1][v] - mn);
        float rs = p0 + p1;
#pragma unroll
        for (int off = 8; off > 0; off >>= 1) rs += __shfl_xor(rs, off, 32);
        lrow[rt][v] = lrow[rt][v] * sc + rs;
        mrow[rt][v] = mn;
        wP[rt][(v + 8 * hf) * 32 + ln]      = (bf16)p0;
        wP[rt][(v + 8 * hf) * 32 + 16 + ln] = (bf16)p1;
#pragma unroll
        for (int nt = 0; nt < 4; ++nt) O[rt][nt][v] *= sc;
      }
    }
    __syncthreads();

    // O += P(16x32) x vT(32x16) for 4 d-tiles
    v16bf vB[4];
#pragma unroll
    for (int nt = 0; nt < 4; ++nt)
      vB[nt] = *(const v16bf*)(vT + (size_t)(bn * DH + nt * 16 + ln) * LCTX + j0 + hf * 16);
#pragma unroll
    for (int rt = 0; rt < 2; ++rt) {
      union { v16bf v; v8bf h[2]; } wa;
      wa.h[0] = *(const v8bf*)(&wP[rt][ln * 32 + hf * 16]);
      wa.h[1] = *(const v8bf*)(&wP[rt][ln * 32 + hf * 16 + 8]);
#pragma unroll
      for (int nt = 0; nt < 4; ++nt)
        O[rt][nt] = wmma_bf16(wa.v, vB[nt], O[rt][nt]);
    }
  }

  // finalize: divide by row sums, write bf16 [B][L][H]
#pragma unroll
  for (int rt = 0; rt < 2; ++rt) {
#pragma unroll
    for (int v = 0; v < 8; ++v) {
      const int row = i0 + rt * 16 + v + 8 * hf;
      const float inv = 1.f / lrow[rt][v];
#pragma unroll
      for (int nt = 0; nt < 4; ++nt)
        attn_out[(size_t)(b * LCTX + row) * HDIM + n * DH + nt * 16 + ln] =
            (bf16)(O[rt][nt][v] * inv);
    }
  }
}

// ---------------------------------------------------------------------------
// Helper kernels
// ---------------------------------------------------------------------------
__global__ void f32_to_bf16_kernel(const float* __restrict__ in,
                                   bf16* __restrict__ out, int nelem) {
  const int gid = blockIdx.x * 256 + threadIdx.x;
  if (gid < nelem) out[gid] = (bf16)in[gid];
}

// W (N,H,D) -> Wt[(n*64+d)][h] bf16 so GEMM B-operand loads are contiguous
__global__ void transpose_nhd_kernel(const float* __restrict__ W,
                                     bf16* __restrict__ Wt) {
  const int idx = blockIdx.x * 256 + threadIdx.x;   // over 512*512
  const int c = idx >> 9, h = idx & 511;
  const int n = c >> 6, d = c & 63;
  Wt[idx] = (bf16)W[((size_t)n * HDIM + h) * DH + d];
}

// ckb[b,n,j] = content_bias[n] . k[b,n,j,:]   prb[n,p] = pos_bias[n] . r[n,p,:]
__global__ void bias_dot_kernel(const bf16* __restrict__ k, const bf16* __restrict__ r,
                                const float* __restrict__ cb, const float* __restrict__ pb,
                                float* __restrict__ ckb, float* __restrict__ prb) {
  const int gid = blockIdx.x * 256 + threadIdx.x;
  const int total_k = NB * NH * LCTX;
  if (gid < total_k) {
    const int bn = gid / LCTX, j = gid % LCTX;
    const int bb = bn >> 3, n = bn & 7;
    const bf16* kp = k + (size_t)(bb * LCTX + j) * HDIM + n * DH;
    const float* cbp = cb + n * DH;
    float s = 0.f;
#pragma unroll
    for (int d = 0; d < DH; ++d) s += (float)kp[d] * cbp[d];
    ckb[gid] = s;
  } else if (gid < total_k + NH * LCTX) {
    const int g = gid - total_k;
    const int n = g / LCTX, p = g % LCTX;
    const bf16* rp = r + (size_t)p * HDIM + n * DH;
    const float* pbp = pb + n * DH;
    float s = 0.f;
#pragma unroll
    for (int d = 0; d < DH; ++d) s += (float)rp[d] * pbp[d];
    prb[g] = s;
  }
}

// out = LayerNorm(A + R) ; optionally also emit bf16 copy for next WMMA GEMM
__global__ __launch_bounds__(128) void ln_residual_kernel(
    const float* __restrict__ A, const float* __restrict__ R,
    const float* __restrict__ w, const float* __restrict__ b,
    float* __restrict__ outF, bf16* __restrict__ outB) {
  const int row = blockIdx.x;
  const int t = threadIdx.x;
  __shared__ float red[128];
  float vals[4];
  float s = 0.f;
#pragma unroll
  for (int i = 0; i < 4; ++i) {
    const int c = t + i * 128;
    const float v = A[(size_t)row * HDIM + c] + R[(size_t)row * HDIM + c];
    vals[i] = v;
    s += v;
  }
  red[t] = s;
  __syncthreads();
  for (int off = 64; off > 0; off >>= 1) {
    if (t < off) red[t] += red[t + off];
    __syncthreads();
  }
  const float mean = red[0] * (1.f / HDIM);
  __syncthreads();
  float s2 = 0.f;
#pragma unroll
  for (int i = 0; i < 4; ++i) {
    const float d = vals[i] - mean;
    s2 += d * d;
  }
  red[t] = s2;
  __syncthreads();
  for (int off = 64; off > 0; off >>= 1) {
    if (t < off) red[t] += red[t + off];
    __syncthreads();
  }
  const float rstd = rsqrtf(red[0] * (1.f / HDIM) + 1e-12f);
#pragma unroll
  for (int i = 0; i < 4; ++i) {
    const int c = t + i * 128;
    const float o = w[c] * (vals[i] - mean) * rstd + b[c];
    outF[(size_t)row * HDIM + c] = o;
    if (outB) outB[(size_t)row * HDIM + c] = (bf16)o;
  }
}

// ---------------------------------------------------------------------------
// Host orchestration
// ---------------------------------------------------------------------------
extern "C" void kernel_launch(void* const* d_in, const int* in_sizes, int n_in,
                              void* d_out, int out_size, void* d_ws, size_t ws_size,
                              hipStream_t stream) {
  (void)in_sizes; (void)n_in; (void)out_size; (void)ws_size;
  const float* x   = (const float*)d_in[0];
  const float* pos = (const float*)d_in[1];
  const float* Wq  = (const float*)d_in[2];
  const float* bq  = (const float*)d_in[3];
  const float* Wk  = (const float*)d_in[4];
  const float* bk  = (const float*)d_in[5];
  const float* Wv  = (const float*)d_in[6];
  const float* bv  = (const float*)d_in[7];
  const float* Wr  = (const float*)d_in[8];
  const float* br  = (const float*)d_in[9];
  const float* cb  = (const float*)d_in[10];
  const float* pb  = (const float*)d_in[11];
  const float* Wc  = (const float*)d_in[12];
  const float* bc  = (const float*)d_in[13];
  const float* W1  = (const float*)d_in[14];
  const float* b1  = (const float*)d_in[15];
  const float* W2  = (const float*)d_in[16];
  const float* b2  = (const float*)d_in[17];
  const float* lnw = (const float*)d_in[18];
  const float* lnb = (const float*)d_in[19];
  float* y = (float*)d_out;

  const int M = NB * LCTX;        // 8192 rows
  char* ws = (char*)d_ws;
  size_t off = 0;
  auto alloc = [&](size_t bytes) -> void* {
    void* p = ws + off;
    off = (off + bytes + 255) & ~(size_t)255;
    return p;
  };

  bf16*  xb   = (bf16*)alloc((size_t)M * HDIM * 2);
  bf16*  posb = (bf16*)alloc((size_t)LCTX * HDIM * 2);
  bf16*  WtQ  = (bf16*)alloc((size_t)HDIM * HDIM * 2);
  bf16*  WtK  = (bf16*)alloc((size_t)HDIM * HDIM * 2);
  bf16*  WtV  = (bf16*)alloc((size_t)HDIM * HDIM * 2);
  bf16*  WtR  = (bf16*)alloc((size_t)HDIM * HDIM * 2);
  bf16*  Wcb  = (bf16*)alloc((size_t)HDIM * HDIM * 2);
  bf16*  W1b  = (bf16*)alloc((size_t)HDIM * HDIM * 2);
  bf16*  W2b  = (bf16*)alloc((size_t)HDIM * HDIM * 2);
  bf16*  qb   = (bf16*)alloc((size_t)M * HDIM * 2);
  bf16*  kbuf = (bf16*)alloc((size_t)M * HDIM * 2);
  bf16*  vT   = (bf16*)alloc((size_t)M * HDIM * 2);
  bf16*  rb   = (bf16*)alloc((size_t)LCTX * HDIM * 2);
  float* ckb  = (float*)alloc((size_t)NB * NH * LCTX * 4);
  float* prb  = (float*)alloc((size_t)NH * LCTX * 4);
  bf16*  attn = (bf16*)alloc((size_t)M * HDIM * 2);
  float* apre = (float*)alloc((size_t)M * HDIM * 4);
  float* af   = (float*)alloc((size_t)M * HDIM * 4);
  bf16*  ab   = (bf16*)alloc((size_t)M * HDIM * 2);
  bf16*  h1   = (bf16*)alloc((size_t)M * HDIM * 2);
  float* h2   = apre;   // apre is dead after first LayerNorm -> reuse

  const int nXH = M * HDIM, nPH = LCTX * HDIM, nWW = HDIM * HDIM;
  f32_to_bf16_kernel<<<(nXH + 255) / 256, 256, 0, stream>>>(x, xb, nXH);
  f32_to_bf16_kernel<<<(nPH + 255) / 256, 256, 0, stream>>>(pos, posb, nPH);
  f32_to_bf16_kernel<<<(nWW + 255) / 256, 256, 0, stream>>>(Wc, Wcb, nWW);
  f32_to_bf16_kernel<<<(nWW + 255) / 256, 256, 0, stream>>>(W1, W1b, nWW);
  f32_to_bf16_kernel<<<(nWW + 255) / 256, 256, 0, stream>>>(W2, W2b, nWW);
  transpose_nhd_kernel<<<(nWW + 255) / 256, 256, 0, stream>>>(Wq, WtQ);
  transpose_nhd_kernel<<<(nWW + 255) / 256, 256, 0, stream>>>(Wk, WtK);
  transpose_nhd_kernel<<<(nWW + 255) / 256, 256, 0, stream>>>(Wv, WtV);
  transpose_nhd_kernel<<<(nWW + 255) / 256, 256, 0, stream>>>(Wr, WtR);

  gemm_bf16_wmma<<<dim3(M / 128, 8), 128, 0, stream>>>(xb, WtQ, bq, qb, M, OUT_BF16);
  gemm_bf16_wmma<<<dim3(M / 128, 8), 128, 0, stream>>>(xb, WtK, bk, kbuf, M, OUT_BF16);
  gemm_bf16_wmma<<<dim3(M / 128, 8), 128, 0, stream>>>(xb, WtV, bv, vT, M, OUT_BF16_VT);
  gemm_bf16_wmma<<<dim3(LCTX / 128, 8), 128, 0, stream>>>(posb, WtR, br, rb, LCTX, OUT_BF16);

  const int nd = NB * NH * LCTX + NH * LCTX;
  bias_dot_kernel<<<(nd + 255) / 256, 256, 0, stream>>>(kbuf, rb, cb, pb, ckb, prb);

  rel_attention_wmma<<<dim3(LCTX / 32, NB * NH), 32, 0, stream>>>(
      qb, kbuf, vT, rb, ckb, prb, attn);

  gemm_bf16_wmma<<<dim3(M / 128, 8), 128, 0, stream>>>(attn, Wcb, bc, apre, M, OUT_F32);
  ln_residual_kernel<<<M, 128, 0, stream>>>(apre, x, lnw, lnb, af, ab);
  gemm_bf16_wmma<<<dim3(M / 128, 8), 128, 0, stream>>>(ab, W1b, b1, h1, M, OUT_GELU_BF16);
  gemm_bf16_wmma<<<dim3(M / 128, 8), 128, 0, stream>>>(h1, W2b, b2, h2, M, OUT_F32);
  ln_residual_kernel<<<M, 128, 0, stream>>>(h2, af, lnw, lnb, y, (bf16*)nullptr);
}
